// _STFTFn_26319559590198
// MI455X (gfx1250) — matrix operand order repaired and checked by reading, not compiled, not run on hardware
//
#include <hip/hip_runtime.h>

// ---- CDNA5 / gfx1250 STFT via f16 WMMA GEMM -------------------------------
// spec = basis(1026x1024) x frames(1024 x 32*1024), then mag/phase.
// Each wave: 16 freqs (real+imag) x 64 frames -> 8 WMMA per 32-wide K step,
// with register double-buffering of the global (B) fragment stream.

typedef __attribute__((ext_vector_type(16))) _Float16     v16h;
typedef __attribute__((ext_vector_type(8)))  float        v8f;
typedef __attribute__((ext_vector_type(4)))  unsigned int u32x4;

#define BATCH    32
#define T_IN     262144
#define LPAD     768                 // win - hop
#define PADT     (T_IN + LPAD)      // 262912
#define HOP      256
#define FILT     1024
#define NFREQ    513
#define NROWS    (2 * NFREQ)        // 1026
#define NFRAMES  1024
#define FTILES   33                 // ceil(513/16)
#define RP       1032               // LDS row pitch in halves (bank-skewed)
#define NT       4                  // frame tiles (16 frames each) per wave

union FragH { v16h h; u32x4 q[2]; };

// ---------------------------------------------------------------- pass 1 ---
__global__ __launch_bounds__(256)
void stft_pad_convert(const float* __restrict__ wav, _Float16* __restrict__ xpad) {
    long long idx = (long long)blockIdx.x * blockDim.x + threadIdx.x;
    const long long total = (long long)BATCH * PADT;
    if (idx >= total) return;
    int b = (int)(idx / PADT);
    int i = (int)(idx % PADT);
    float v = (i < LPAD) ? 0.0f : wav[(long long)b * T_IN + (i - LPAD)];
    xpad[idx] = (_Float16)v;
}

// ---------------------------------------------------------------- pass 2 ---
__global__ __launch_bounds__(256)
void stft_basis_convert(const float* __restrict__ basis, _Float16* __restrict__ bf16) {
    long long idx = (long long)blockIdx.x * blockDim.x + threadIdx.x;
    const long long total = (long long)NROWS * FILT;
    if (idx >= total) return;
    bf16[idx] = (_Float16)basis[idx];
}

// ----------------------------------------------------- fragment helpers ----
__device__ __forceinline__ void load_B(const _Float16* const bptr[NT], int k0,
                                       FragH* bB) {
    #pragma unroll
    for (int j = 0; j < NT; ++j) {
        // B fragment: halves 0..7 = K[k0+bk .. +7], halves 8..15 = K[+16 .. +23]
        bB[j].q[0] = *(const u32x4*)(bptr[j] + k0);
        bB[j].q[1] = *(const u32x4*)(bptr[j] + k0 + 16);
    }
}

__device__ __forceinline__ void load_A(const _Float16* are_base,
                                       const _Float16* aim_base, int k0,
                                       FragH& a_re, FragH& a_im) {
    // A fragment: halves 0..7 = K[k0+asel .. +7], halves 8..15 = K[+16 .. +23]
    a_re.q[0] = *(const u32x4*)(are_base + k0);
    a_re.q[1] = *(const u32x4*)(are_base + k0 + 16);
    a_im.q[0] = *(const u32x4*)(aim_base + k0);
    a_im.q[1] = *(const u32x4*)(aim_base + k0 + 16);
}

__device__ __forceinline__ void wmma8(const FragH& a_re, const FragH& a_im,
                                      const FragH* bB, v8f* acc_re, v8f* acc_im) {
    #pragma unroll
    for (int j = 0; j < NT; ++j) {
        acc_re[j] = __builtin_amdgcn_wmma_f32_16x16x32_f16(
            false, a_re.h, false, bB[j].h, (short)0, acc_re[j], false, false);
        acc_im[j] = __builtin_amdgcn_wmma_f32_16x16x32_f16(
            false, a_im.h, false, bB[j].h, (short)0, acc_im[j], false, false);
    }
}

// ---------------------------------------------------------------- pass 3 ---
// block: 8 waves. blockIdx.y = freq tile (16 freqs, real+imag), blockIdx.z = batch,
// wave w handles frames [ (blockIdx.x*8+w)*64 , +64 ) as 4 WMMA N-tiles.
__global__ __launch_bounds__(256)
void stft_wmma_gemm(const _Float16* __restrict__ xpad,
                    const _Float16* __restrict__ basis,
                    float* __restrict__ out) {
    extern __shared__ _Float16 sb[];           // [2][16][RP] = 66048 B

    const int tid   = threadIdx.x;
    const int lane  = tid & 31;
    const int wave  = tid >> 5;
    const int b     = blockIdx.z;
    const int fbase = blockIdx.y * 16;
    const int t0    = (blockIdx.x * 8 + wave) * (16 * NT);

    // ---- cooperatively stage basis tile (16 real rows + 16 imag rows) ----
    #pragma unroll
    for (int it = 0; it < 16; ++it) {
        int idx   = it * 256 + tid;
        int cc    = idx & 127;                 // 8-half chunk within row
        int rf    = idx >> 7;                  // 0..31
        int plane = rf >> 4;
        int r     = rf & 15;
        int srow  = plane ? (NFREQ + fbase + r) : (fbase + r);
        if (srow > NROWS - 1) srow = NROWS - 1;   // clamp freq-tile padding
        const u32x4 v = *(const u32x4*)(basis + (long long)srow * FILT + cc * 8);
        *(u32x4*)(sb + plane * 16 * RP + r * RP + cc * 8) = v;
    }
    __syncthreads();

    // ---- per-lane fragment addressing (ISA 7.12.2 layouts, wave32) ----
    const int arow = lane & 15;                // A: M = lane%16
    const int asel = (lane >> 4) * 8;          // A: lane[4] selects K sub-block
    const _Float16* are_base = sb + arow * RP + asel;
    const _Float16* aim_base = sb + 16 * RP + arow * RP + asel;

    const int bcol = lane & 15;                // B: N = lane%16 (frame)
    const int bk   = (lane >> 4) * 8;          // B: lane[4] selects K sub-block
    const _Float16* bptr[NT];
    #pragma unroll
    for (int j = 0; j < NT; ++j) {
        bptr[j] = xpad + (long long)b * PADT
                       + (long long)(t0 + j * 16 + bcol) * HOP + bk;
        __builtin_prefetch(bptr[j], 0, 1);     // global_prefetch_b8
    }

    v8f acc_re[NT];
    v8f acc_im[NT];
    #pragma unroll
    for (int j = 0; j < NT; ++j) { acc_re[j] = (v8f){}; acc_im[j] = (v8f){}; }

    // ---- K loop: ping-pong B fragments so next-step loads overlap WMMAs ----
    FragH B0[NT], B1[NT], aF_re, aF_im;
    load_B(bptr, 0, B0);
    for (int k0 = 0; k0 < FILT - 64; k0 += 64) {
        load_B(bptr, k0 + 32, B1);                 // in flight during wmma(B0)
        load_A(are_base, aim_base, k0, aF_re, aF_im);
        wmma8(aF_re, aF_im, B0, acc_re, acc_im);
        load_B(bptr, k0 + 64, B0);                 // in flight during wmma(B1)
        load_A(are_base, aim_base, k0 + 32, aF_re, aF_im);
        wmma8(aF_re, aF_im, B1, acc_re, acc_im);
    }
    // tail: k-steps FILT-64 and FILT-32 (B0 already holds FILT-64)
    load_B(bptr, FILT - 32, B1);
    load_A(are_base, aim_base, FILT - 64, aF_re, aF_im);
    wmma8(aF_re, aF_im, B0, acc_re, acc_im);
    load_A(are_base, aim_base, FILT - 32, aF_re, aF_im);
    wmma8(aF_re, aF_im, B1, acc_re, acc_im);

    // ---- epilogue: C/D layout -> magnitude & phase ----
    // VGPR r: lanes 0-15 -> (M=r,   N=lane), lanes 16-31 -> (M=r+8, N=lane-16)
    const int M0 = (lane >> 4) * 8;
    const int N  = lane & 15;
    const long long magsz = (long long)BATCH * NFREQ * NFRAMES;

    #pragma unroll
    for (int j = 0; j < NT; ++j) {
        const int t = t0 + j * 16 + N;
        #pragma unroll
        for (int r = 0; r < 8; ++r) {
            int f = fbase + M0 + r;
            if (f < NFREQ) {
                float re  = acc_re[j][r];
                float im  = acc_im[j][r];
                float mag = sqrtf(re * re + im * im);
                float ph  = atan2f(im, re);
                long long o = ((long long)b * NFREQ + f) * NFRAMES + t;
                out[o]         = mag;
                out[magsz + o] = ph;
            }
        }
    }
}

// --------------------------------------------------------------- launcher --
extern "C" void kernel_launch(void* const* d_in, const int* in_sizes, int n_in,
                              void* d_out, int out_size, void* d_ws, size_t ws_size,
                              hipStream_t stream) {
    const float* wav   = (const float*)d_in[0];   // (32, 262144) f32
    const float* basis = (const float*)d_in[1];   // (1026, 1, 1024) f32
    // d_in[2]=hop(256), d_in[3]=win(1024): compile-time constants here.

    _Float16* xpad = (_Float16*)d_ws;                                   // 16.83 MB
    _Float16* bf16 = (_Float16*)((char*)d_ws +
                         (size_t)BATCH * PADT * sizeof(_Float16));      // +2.10 MB
    float* out = (float*)d_out;

    {
        long long total = (long long)BATCH * PADT;
        int blocks = (int)((total + 255) / 256);
        stft_pad_convert<<<blocks, 256, 0, stream>>>(wav, xpad);
    }
    {
        long long total = (long long)NROWS * FILT;
        int blocks = (int)((total + 255) / 256);
        stft_basis_convert<<<blocks, 256, 0, stream>>>(basis, bf16);
    }
    {
        dim3 grid(NFRAMES / (128 * NT), FTILES, BATCH);   // (2, 33, 32)
        size_t shmem = (size_t)2 * 16 * RP * sizeof(_Float16);  // 66048 B
        stft_wmma_gemm<<<grid, 256, shmem, stream>>>(xpad, bf16, out);
    }
}
